// _ElmoBiLm_57595511439703
// MI455X (gfx1250) — matrix operand order repaired and checked by reading, not compiled, not run on hardware
//
#include <hip/hip_runtime.h>
#include <hip/hip_bf16.h>

typedef __bf16 v16bf __attribute__((ext_vector_type(16)));
typedef float  v8f   __attribute__((ext_vector_type(8)));

#define NLAYERS 2
#define ISZ  512
#define HID  512
#define CELL 4096
#define GATES (4*CELL)        // 16384
#define BATCH 32
#define SEQ   256
#define KDIM  1024            // ISZ + HID
#define OFEAT 1024            // 2*HID
#define OS    (BATCH*SEQ*OFEAT) // 8388608 elems per output slab

// ---- helpers -------------------------------------------------------------
__device__ __forceinline__ __bf16 f2bf(float f) {
    union { float f; unsigned u; } in; in.f = f;
    unsigned u = in.u;
    u += 0x7fffu + ((u >> 16) & 1u);          // round-to-nearest-even
    union { unsigned short s; __bf16 b; } out;
    out.s = (unsigned short)(u >> 16);
    return out.b;
}
__device__ __forceinline__ float sigmoidf(float x) {
    return 1.0f / (1.0f + __expf(-x));
}
__device__ __forceinline__ float clip3(float x) {
    return fminf(3.0f, fmaxf(-3.0f, x));
}
// A-fragment K order for 16-bit A 16x32: contiguous k = 16*(lane/16) + i (matches B packing)
__device__ __forceinline__ unsigned a_k_of(unsigned lane, unsigned i) {
    unsigned hi = lane >> 4;
    return hi*16 + i;
}

// ---- weight conversion + packing (fp32 -> bf16 WMMA A-fragments) ---------
__global__ void k_pack_weights(const float* __restrict__ Wi_f, const float* __restrict__ Ws_f,
                               const float* __restrict__ Wp_f, const float* __restrict__ Wi_b,
                               const float* __restrict__ Ws_b, const float* __restrict__ Wp_b,
                               __bf16* __restrict__ wpack, __bf16* __restrict__ wppack) {
    unsigned tid = blockIdx.x * 256u + threadIdx.x;
    const unsigned NW = 4u * GATES * KDIM;      // 67,108,864
    if (tid < NW) {
        unsigned e = tid;
        unsigned ld   = e / (GATES * KDIM);
        unsigned rem  = e % (GATES * KDIM);
        unsigned frag = rem >> 9, within = rem & 511;
        unsigned lane = within >> 4, i = within & 15;
        unsigned mtile = frag >> 5, ktile = frag & 31;       // 32 ktiles (K=1024)
        unsigned M = mtile * 16 + (lane & 15);
        unsigned K = ktile * 32 + a_k_of(lane, i);
        unsigned l = ld >> 1, d = ld & 1;
        const float* Wi = d ? Wi_b : Wi_f;
        const float* Ws = d ? Ws_b : Ws_f;
        float val = (K < ISZ) ? Wi[((size_t)l*GATES + M)*ISZ + K]
                              : Ws[((size_t)l*GATES + M)*HID + (K - ISZ)];
        wpack[e] = f2bf(val);
    } else {
        unsigned e = tid - NW;
        if (e >= 4u * HID * CELL) return;
        unsigned per  = HID * CELL;             // 2,097,152
        unsigned ld   = e / per, rem = e % per;
        unsigned frag = rem >> 9, within = rem & 511;
        unsigned lane = within >> 4, i = within & 15;
        unsigned mtile = frag >> 7, ktile = frag & 127;      // 128 ktiles (K=4096)
        unsigned M = mtile * 16 + (lane & 15);
        unsigned K = ktile * 32 + a_k_of(lane, i);
        unsigned l = ld >> 1, d = ld & 1;
        const float* Wp = d ? Wp_b : Wp_f;
        wppack[e] = f2bf(Wp[((size_t)l*HID + M)*CELL + K]);
    }
}

// ---- d_out[0] = concat(inputs, inputs) -----------------------------------
__global__ void k_init_out0(const float* __restrict__ x, float* __restrict__ out) {
    int tid = blockIdx.x * 256 + threadIdx.x;
    if (tid >= OS) return;
    int b = tid / (SEQ * OFEAT);
    int r = tid % (SEQ * OFEAT);
    int t = r / OFEAT, c = r % OFEAT;
    out[tid] = x[((size_t)b * SEQ + t) * ISZ + (c & (ISZ - 1))];
}

// ---- zero h,c state at layer start ---------------------------------------
__global__ void k_zero_state(float* __restrict__ c, float* __restrict__ h) {
    int tid = blockIdx.x * 256 + threadIdx.x;
    const int NC = 2 * BATCH * CELL;            // 262144
    const int NH = 2 * BATCH * HID;             // 32768
    if (tid < NC) c[tid] = 0.0f;
    else if (tid < NC + NH) h[tid - NC] = 0.0f;
}

// ---- pack z = [x_t ; h] into bf16 B-fragments ----------------------------
__global__ void k_pack_z(int l, int t, const float* __restrict__ in0,
                         const float* __restrict__ out, const float* __restrict__ h,
                         v16bf* __restrict__ zfrag) {
    int tid  = blockIdx.x * 256 + threadIdx.x;  // 4096 threads
    int lane = tid & 31;
    int nt   = (tid >> 5) & 1;
    int kt   = (tid >> 6) & 31;
    int dir  = tid >> 11;
    int b     = nt * 16 + (lane & 15);
    int kbase = kt * 32 + (lane >> 4) * 16;
    int td    = dir ? (SEQ - 1 - t) : t;
    v16bf r;
#pragma unroll
    for (int i = 0; i < 16; ++i) {
        int k = kbase + i;
        float val;
        if (k < ISZ) {
            val = (l == 0) ? in0[((size_t)b * SEQ + td) * ISZ + k]
                           : out[(size_t)l * OS + ((size_t)b * SEQ + td) * OFEAT + dir * HID + k];
        } else {
            val = h[((size_t)dir * BATCH + b) * HID + (k - ISZ)];
        }
        r[i] = f2bf(val);
    }
    zfrag[((dir * 32 + kt) * 2 + nt) * 32 + lane] = r;
}

// ---- gates^T(16384x32) = [Wi|Ws](bf16) x z(bf16), f32 accum --------------
__global__ void k_gates(int l, const v16bf* __restrict__ wpack,
                        const v16bf* __restrict__ zfrag, float* __restrict__ gatesT) {
    int lane  = threadIdx.x & 31;
    int w     = blockIdx.x * 8 + (threadIdx.x >> 5);   // 2048 waves
    int dir   = w >> 10;
    int mtile = w & 1023;
    const v16bf* wp = wpack + ((size_t)(l * 2 + dir) * 1024 * 32 + (size_t)mtile * 32) * 32;
    const v16bf* zf = zfrag + (size_t)dir * (32 * 2 * 32);
    v8f a0 = {}; v8f a1 = {};
    for (int kt = 0; kt < 32; ++kt) {
        v16bf a  = wp[kt * 32 + lane];
        v16bf b0 = zf[(kt * 2 + 0) * 32 + lane];
        v16bf b1 = zf[(kt * 2 + 1) * 32 + lane];
        a0 = __builtin_amdgcn_wmma_f32_16x16x32_bf16(false, a, false, b0, (short)0, a0, false, false);
        a1 = __builtin_amdgcn_wmma_f32_16x16x32_bf16(false, a, false, b1, (short)0, a1, false, false);
    }
    float* g = gatesT + (size_t)dir * GATES * BATCH;
    int hi = lane >> 4, col = lane & 15;
#pragma unroll
    for (int v = 0; v < 8; ++v) {
        int row = mtile * 16 + v + 8 * hi;
        g[row * BATCH + col]      = a0[v];
        g[row * BATCH + 16 + col] = a1[v];
    }
}

// ---- elementwise LSTM cell update ----------------------------------------
__global__ void k_cell(int l, int t, const float* __restrict__ gatesT,
                       const float* __restrict__ b_f, const float* __restrict__ b_b,
                       const int* __restrict__ len, float* __restrict__ cst,
                       float* __restrict__ pre) {
    int tid = blockIdx.x * 256 + threadIdx.x;          // 262144 threads
    int dir = tid >> 17;
    int rem = tid & 131071;
    int b   = rem >> 12;
    int j   = rem & 4095;
    const float* g    = gatesT + (size_t)dir * GATES * BATCH;
    const float* bias = (dir ? b_b : b_f) + (size_t)l * GATES;
    float gi = g[(size_t)(0 * CELL + j) * BATCH + b] + bias[j];
    float gf = g[(size_t)(1 * CELL + j) * BATCH + b] + bias[CELL + j];
    float gg = g[(size_t)(2 * CELL + j) * BATCH + b] + bias[2 * CELL + j];
    float go = g[(size_t)(3 * CELL + j) * BATCH + b] + bias[3 * CELL + j];
    float ig = sigmoidf(gi), fg = sigmoidf(gf), og = sigmoidf(go), gv = tanhf(gg);
    size_t ci = ((size_t)dir * BATCH + b) * CELL + j;
    float cold = cst[ci];
    float cn = clip3(ig * gv + fg * cold);
    int td = dir ? (SEQ - 1 - t) : t;
    bool m = td < len[b];
    cst[ci] = m ? cn : cold;
    pre[ci] = og * tanhf(cn);
}

// ---- projection h^T(512x32) = Wp(bf16) x pre^T, clip, mask, write out ----
__global__ void k_proj(int l, int t, const v16bf* __restrict__ wppack,
                       const float* __restrict__ pre, const int* __restrict__ len,
                       float* __restrict__ h, float* __restrict__ out) {
    int lane  = threadIdx.x & 31;
    int w     = blockIdx.x * 8 + (threadIdx.x >> 5);   // 64 waves
    int dir   = w >> 5;
    int mtile = w & 31;
    const v16bf* wp = wppack + ((size_t)(l * 2 + dir) * 32 * 128 + (size_t)mtile * 128) * 32;
    int hi = lane >> 4, cl = lane & 15;
    v8f a0 = {}; v8f a1 = {};
    for (int kt = 0; kt < 128; ++kt) {
        v16bf a = wp[kt * 32 + lane];
        int kbase = kt * 32 + hi * 16;
        const float* p0 = pre + ((size_t)dir * BATCH + cl) * CELL + kbase;
        const float* p1 = pre + ((size_t)dir * BATCH + 16 + cl) * CELL + kbase;
        v16bf b0, b1;
#pragma unroll
        for (int i = 0; i < 16; ++i) { b0[i] = f2bf(p0[i]); b1[i] = f2bf(p1[i]); }
        a0 = __builtin_amdgcn_wmma_f32_16x16x32_bf16(false, a, false, b0, (short)0, a0, false, false);
        a1 = __builtin_amdgcn_wmma_f32_16x16x32_bf16(false, a, false, b1, (short)0, a1, false, false);
    }
    int td = dir ? (SEQ - 1 - t) : t;
#pragma unroll
    for (int v = 0; v < 8; ++v) {
        int row = mtile * 16 + v + 8 * hi;
#pragma unroll
        for (int nt = 0; nt < 2; ++nt) {
            int b = nt * 16 + cl;
            float hv = clip3(nt ? a1[v] : a0[v]);
            bool m = td < len[b];
            size_t hidx = ((size_t)dir * BATCH + b) * HID + row;
            float hold = h[hidx];
            h[hidx] = m ? hv : hold;
            size_t base = ((size_t)b * SEQ + td) * OFEAT + (size_t)dir * HID + row;
            float resid = l ? out[(size_t)l * OS + base] : 0.0f;
            out[(size_t)(l + 1) * OS + base] = (m ? hv : 0.0f) + resid;
        }
    }
}

// ---- host launcher -------------------------------------------------------
extern "C" void kernel_launch(void* const* d_in, const int* in_sizes, int n_in,
                              void* d_out, int out_size, void* d_ws, size_t ws_size,
                              hipStream_t stream) {
    (void)in_sizes; (void)n_in; (void)out_size; (void)ws_size;
    const float* inputs  = (const float*)d_in[0];
    const int*   lengths = (const int*)d_in[1];
    const float* Wi_f = (const float*)d_in[2];
    const float* Ws_f = (const float*)d_in[3];
    const float* b_f  = (const float*)d_in[4];
    const float* Wp_f = (const float*)d_in[5];
    const float* Wi_b = (const float*)d_in[6];
    const float* Ws_b = (const float*)d_in[7];
    const float* b_b  = (const float*)d_in[8];
    const float* Wp_b = (const float*)d_in[9];
    float* out = (float*)d_out;

    char* ws = (char*)d_ws;
    constexpr size_t WPACK_B  = 4ull * GATES * KDIM * 2;   // 134,217,728
    constexpr size_t WPPACK_B = 4ull * HID * CELL * 2;     //  16,777,216
    constexpr size_t ZFRAG_B  = 2ull * 2048 * 32;          //     131,072
    constexpr size_t GATES_B  = 2ull * GATES * BATCH * 4;  //   4,194,304
    constexpr size_t PRE_B    = 2ull * BATCH * CELL * 4;   //   1,048,576
    constexpr size_t C_B      = PRE_B;
    __bf16* wpack  = (__bf16*)(ws);
    __bf16* wppack = (__bf16*)(ws + WPACK_B);
    v16bf*  zfrag  = (v16bf*)(ws + WPACK_B + WPPACK_B);
    float*  gatesT = (float*)(ws + WPACK_B + WPPACK_B + ZFRAG_B);
    float*  pre    = (float*)(ws + WPACK_B + WPPACK_B + ZFRAG_B + GATES_B);
    float*  cst    = (float*)(ws + WPACK_B + WPPACK_B + ZFRAG_B + GATES_B + PRE_B);
    float*  hst    = (float*)(ws + WPACK_B + WPPACK_B + ZFRAG_B + GATES_B + PRE_B + C_B);

    const v16bf* wpack_v  = (const v16bf*)wpack;
    const v16bf* wppack_v = (const v16bf*)wppack;

    // one-time per call: bf16-pack weights (resident in 192MB L2 afterwards)
    k_pack_weights<<<294912, 256, 0, stream>>>(Wi_f, Ws_f, Wp_f, Wi_b, Ws_b, Wp_b, wpack, wppack);
    k_init_out0<<<OS / 256, 256, 0, stream>>>(inputs, out);

    for (int l = 0; l < NLAYERS; ++l) {
        k_zero_state<<<1152, 256, 0, stream>>>(cst, hst);
        for (int t = 0; t < SEQ; ++t) {
            k_pack_z<<<16,   256, 0, stream>>>(l, t, inputs, out, hst, zfrag);
            k_gates<<<256,   256, 0, stream>>>(l, wpack_v, zfrag, gatesT);
            k_cell<<<1024,   256, 0, stream>>>(l, t, gatesT, b_f, b_b, lengths, cst, pre);
            k_proj<<<8,      256, 0, stream>>>(l, t, wppack_v, pre, lengths, hst, out);
        }
    }
}
